// MinkEncoderDecoder_40699110096958
// MI455X (gfx1250) — matrix operand order repaired and checked by reading, not compiled, not run on hardware
//
#include <hip/hip_runtime.h>

#define N0V   50000
#define N1V   15000
#define TCS   64      // K-chunk staged in LDS per iteration (2 WMMA steps)

typedef __attribute__((ext_vector_type(16))) __bf16       v16bf;
typedef __attribute__((ext_vector_type(8)))  float        v8f;
typedef __attribute__((ext_vector_type(4)))  float        v4f;
typedef __attribute__((ext_vector_type(4)))  unsigned int v4u;

// ---------------------------------------------------------------------------
// Generic gather-GEMM sparse conv:  out[n,co] = sum_{k,ci} feat[nbr[n,k],ci] * W[k*Ci+ci, co]
// nbr == nullptr -> identity gather (dense GEMM: 1x1 downsample & head).
// 8 waves / block; block tile = 32 rows x 64 cols; wave (r,c) in 2x4 grid owns
// one 16x16 WMMA tile. Requires Ci%4==0, Co%4==0 (true for every layer here),
// so staging uses 16B global loads and an exact magic-multiply for l/Ci.
// ---------------------------------------------------------------------------
__global__ __launch_bounds__(256)
void sconv_wmma_bf16(const float* __restrict__ feat,
                     const int*   __restrict__ nbr,
                     const float* __restrict__ W,
                     const float* __restrict__ bias,
                     float*       __restrict__ out,
                     int Nout, int K, int Ci, int Co, int ciInv) {
  __shared__ __bf16 As[32][TCS];        // gathered activations, bf16
  __shared__ __bf16 Bs[64][TCS + 8];    // weights, transposed [col][k], +8 pad vs bank conflicts

  const int tid   = threadIdx.x;
  const int wave  = tid >> 5;
  const int lane  = tid & 31;
  const int wrow  = wave >> 2;          // 0..1  (row tile within block)
  const int wcol  = wave & 3;           // 0..3  (col tile within block)
  const int h     = (lane >> 4) & 1;    // lane half (ISA 16x16 layout)
  const int n     = lane & 15;
  const int rb0   = blockIdx.x << 5;    // block row base (32 rows)
  const int cob0  = blockIdx.y << 6;    // block col base (64 cols)
  const int col   = cob0 + (wcol << 4) + n;
  const int L     = K * Ci;

  v8f acc = {};

  for (int l0 = 0; l0 < L; l0 += TCS) {
    // ---- stage A: 32 rows x TCS gathered features, 4 elems (16B) per thread-iter
    for (int idx = tid; idx < 32 * (TCS / 4); idx += 256) {
      int rr  = idx >> 4;               // row within tile
      int cc  = (idx & 15) << 2;        // 4-aligned column in chunk
      int l   = l0 + cc;
      int row = rb0 + rr;
      union { __bf16 b[4]; unsigned long long u; } pk;
      pk.u = 0ull;
      if (l < L && row < Nout) {        // 4-group never crosses a tap (Ci%4==0)
        unsigned tap = ((unsigned)l * (unsigned)ciInv) >> 20;   // exact l/Ci
        int ch  = l - (int)tap * Ci;
        int src = nbr ? nbr[(size_t)row * K + tap] : row;
        if (src >= 0) {
          v4f v = *(const v4f*)(feat + (size_t)src * Ci + ch);  // 16B aligned
          pk.b[0] = (__bf16)v.x; pk.b[1] = (__bf16)v.y;
          pk.b[2] = (__bf16)v.z; pk.b[3] = (__bf16)v.w;
        }
      }
      *(unsigned long long*)&As[rr][cc] = pk.u;
    }
    // ---- stage B (transposed): Bs[col][k] = W[(l0+k)*Co + cob0+col], 16B reads
    for (int idx = tid; idx < TCS * 16; idx += 256) {
      int rr = idx >> 4;                // k within chunk
      int cc = (idx & 15) << 2;         // 4-aligned col within group
      int l  = l0 + rr;
      int co = cob0 + cc;
      v4f v = {};
      if (l < L && co < Co) v = *(const v4f*)(W + (size_t)l * Co + co);  // Co%4==0
      Bs[cc + 0][rr] = (__bf16)v.x;
      Bs[cc + 1][rr] = (__bf16)v.y;
      Bs[cc + 2][rr] = (__bf16)v.z;
      Bs[cc + 3][rr] = (__bf16)v.w;
    }
    __syncthreads();

#pragma unroll
    for (int kk = 0; kk < TCS; kk += 32) {
      // A fragment: lane(h, m=n): e0..7 <- K = h*8+0..7 ; e8..15 <- K = 16+h*8+0..7
      union { v16bf v; v4u u[2]; } a;
      const __bf16* pa = &As[(wrow << 4) + n][kk + (h << 3)];
      a.u[0] = *(const v4u*)pa;
      a.u[1] = *(const v4u*)(pa + 16);
      // B fragment: lane(h, n): e0..7 <- K = h*8+0..7 ; e8..15 <- K = 16+h*8+0..7
      union { v16bf v; v4u u[2]; } b;
      const __bf16* pb = &Bs[(wcol << 4) + n][kk + (h << 3)];
      b.u[0] = *(const v4u*)pb;
      b.u[1] = *(const v4u*)(pb + 16);

      acc = __builtin_amdgcn_wmma_f32_16x16x32_bf16(false, a.v, false, b.v,
                                                    (short)0, acc, false, false);
    }
    __syncthreads();
  }

  // D layout: VGPR r -> M = r + 8*h, N = lane&15
  if (col < Co) {
    float bi = bias ? bias[col] : 0.f;
    const int rowb = rb0 + (wrow << 4) + (h << 3);
#pragma unroll
    for (int r = 0; r < 8; ++r) {
      int row = rowb + r;
      if (row < Nout) out[(size_t)row * Co + col] = acc[r] + bi;
    }
  }
}

// ---------------------------------------------------------------------------
// Per-channel mean & rsqrt(var+eps) over rows (training-mode BN stats)
// ---------------------------------------------------------------------------
__global__ void colstats_kernel(const float* __restrict__ x, int N, int C,
                                float* __restrict__ mean, float* __restrict__ rsig) {
  int c = blockIdx.x;
  float s = 0.f, q = 0.f;
  for (int r = threadIdx.x; r < N; r += blockDim.x) {
    float v = x[(size_t)r * C + c];
    s += v; q += v * v;
  }
  __shared__ float sh_s[256], sh_q[256];
  sh_s[threadIdx.x] = s; sh_q[threadIdx.x] = q;
  __syncthreads();
  for (int w = 128; w > 0; w >>= 1) {
    if (threadIdx.x < (unsigned)w) {
      sh_s[threadIdx.x] += sh_s[threadIdx.x + w];
      sh_q[threadIdx.x] += sh_q[threadIdx.x + w];
    }
    __syncthreads();
  }
  if (threadIdx.x == 0) {
    float m   = sh_s[0] / (float)N;
    float var = sh_q[0] / (float)N - m * m;
    mean[c] = m;
    rsig[c] = rsqrtf(var + 1e-5f);
  }
}

// ---------------------------------------------------------------------------
// Fused BN affine + optional skip-add + activation + strided store.
// mean==nullptr -> plain copy path (used for the concat copy).
// act: 0 none, 1 relu, 2 leaky-relu(0.01)
// ---------------------------------------------------------------------------
__global__ void bn_act_kernel(const float* __restrict__ x,
                              const float* __restrict__ mean, const float* __restrict__ rsig,
                              const float* __restrict__ g, const float* __restrict__ b,
                              const float* __restrict__ skip,
                              float* __restrict__ out,
                              int N, int C, int ostride, int ocol, int act) {
  int i = blockIdx.x * blockDim.x + threadIdx.x;
  if (i >= N * C) return;
  int r = i / C, c = i - r * C;
  float v = x[i];
  if (mean) v = (v - mean[c]) * rsig[c] * g[c] + b[c];
  if (skip) v += skip[i];
  if (act == 1)      v = fmaxf(v, 0.f);
  else if (act == 2) v = (v > 0.f) ? v : 0.01f * v;
  out[(size_t)r * ostride + ocol + c] = v;
}

// ---------------------------------------------------------------------------
extern "C" void kernel_launch(void* const* d_in, const int* in_sizes, int n_in,
                              void* d_out, int out_size, void* d_ws, size_t ws_size,
                              hipStream_t stream) {
  (void)in_sizes; (void)n_in; (void)out_size; (void)ws_size;

  // Inputs: recursive insertion-order flatten of setup_inputs() dict.
  const float* feats = (const float*)d_in[0];
  const float* P[49];
  for (int i = 0; i < 49; ++i) P[i] = (const float*)d_in[i];
  const int* nbr_full   = (const int*)d_in[45];
  const int* nbr_down   = (const int*)d_in[46];
  const int* nbr_coarse = (const int*)d_in[47];
  const int* nbr_up     = (const int*)d_in[48];

  float* ws = (float*)d_ws;
  size_t off = 0;
  auto wsalloc = [&](size_t nf) { float* p = ws + off; off += (nf + 63) & ~(size_t)63; return p; };

  float* cat  = wsalloc((size_t)N0V * 128);  // concat buf; also hosts coarse bufs & res3-conv2 tmp
  float* tf   = wsalloc((size_t)N0V * 96);   // full-res conv tmp
  float* uf   = wsalloc((size_t)N0V * 96);   // res mid; aliases stem/ds tmp
  float* sf   = wsalloc((size_t)N0V * 96);   // full-res skip; aliases stem x0a
  float* x0   = wsalloc((size_t)N0V * 32);   // stem output (kept for concat)
  float* mean = wsalloc(128);
  float* rsig = wsalloc(128);

  float* tA  = uf;                               // [N0,32] stem tmp / [N1,32] ds tmp
  float* x0a = sf;                               // [N0,32]
  float* tc  = cat;                              // [N1,64] coarse conv tmp
  float* xc1 = cat + (size_t)N1V * 64;           // [N1,64]
  float* xc2 = cat + (size_t)N1V * 64 * 2;       // [N1,64]
  float* sc  = cat + (size_t)N1V * 64 * 3;       // [N1,64] coarse skip
  float* t2  = cat;                              // [N0,96] res3 conv2 tmp (cat dead by then)

  float* yout   = (float*)d_out;                 // [N0,96]
  float* logits = (float*)d_out + (size_t)N0V * 96;

  auto conv = [&](const float* f, const int* nb, const float* Wt, const float* bias,
                  float* o, int Nout, int K, int Ci, int Co) {
    int ciInv = (1 << 20) / Ci + 1;              // exact l/Ci for l*Ci < 2^20
    dim3 g((Nout + 31) / 32, (Co + 63) / 64);
    sconv_wmma_bf16<<<g, dim3(256), 0, stream>>>(f, nb, Wt, bias, o, Nout, K, Ci, Co, ciInv);
  };
  auto bn = [&](const float* x, const float* g_, const float* b_, const float* skip,
                float* o, int N, int C, int ostride, int ocol, int act) {
    colstats_kernel<<<dim3(C), dim3(256), 0, stream>>>(x, N, C, mean, rsig);
    int tot = N * C;
    bn_act_kernel<<<dim3((tot + 255) / 256), dim3(256), 0, stream>>>(
        x, mean, rsig, g_, b_, skip, o, N, C, ostride, ocol, act);
  };

  // ---- stem: two ks=3 convs at full res
  conv(feats, nbr_full, P[1], nullptr, tA, N0V, 27, 4, 32);
  bn(tA, P[2], P[3], nullptr, x0a, N0V, 32, 32, 0, 1);
  conv(x0a, nbr_full, P[4], nullptr, tA, N0V, 27, 32, 32);
  bn(tA, P[5], P[6], nullptr, x0, N0V, 32, 32, 0, 1);

  // ---- downsample (ks=2 stride-2) + lrelu
  conv(x0, nbr_down, P[7], nullptr, tA, N1V, 8, 32, 32);
  bn(tA, P[8], P[9], nullptr, xc1, N1V, 32, 32, 0, 2);

  // ---- res1 (32 -> 64, with 1x1 downsample skip)
  conv(xc1, nbr_coarse, P[10], nullptr, tc, N1V, 27, 32, 64);
  bn(tc, P[11], P[12], nullptr, xc2, N1V, 64, 64, 0, 1);
  conv(xc2, nbr_coarse, P[13], nullptr, tc, N1V, 27, 64, 64);
  conv(xc1, nullptr, P[16], nullptr, sc, N1V, 1, 32, 64);      // skip GEMM
  bn(sc, P[17], P[18], nullptr, sc, N1V, 64, 64, 0, 0);        // skip BN (in-place)
  bn(tc, P[14], P[15], sc, xc1, N1V, 64, 64, 0, 1);            // BN + skip + relu

  // ---- res2 (64 -> 64, identity skip)
  conv(xc1, nbr_coarse, P[19], nullptr, tc, N1V, 27, 64, 64);
  bn(tc, P[20], P[21], nullptr, xc2, N1V, 64, 64, 0, 1);
  conv(xc2, nbr_coarse, P[22], nullptr, tc, N1V, 27, 64, 64);
  bn(tc, P[23], P[24], xc1, xc2, N1V, 64, 64, 0, 1);

  // ---- up (ks=2 transposed) + lrelu into cat[:,0:96], concat x0 into cat[:,96:128]
  conv(xc2, nbr_up, P[25], nullptr, tf, N0V, 8, 64, 96);
  bn(tf, P[26], P[27], nullptr, cat, N0V, 96, 128, 0, 2);
  {
    int tot = N0V * 32;
    bn_act_kernel<<<dim3((tot + 255) / 256), dim3(256), 0, stream>>>(
        x0, nullptr, nullptr, nullptr, nullptr, nullptr, cat, N0V, 32, 128, 96, 0);
  }

  // ---- res3 (128 -> 96, with 1x1 downsample skip)
  conv(cat, nullptr, P[34], nullptr, tf, N0V, 1, 128, 96);     // skip GEMM
  bn(tf, P[35], P[36], nullptr, sf, N0V, 96, 96, 0, 0);
  conv(cat, nbr_full, P[28], nullptr, tf, N0V, 27, 128, 96);
  bn(tf, P[29], P[30], nullptr, uf, N0V, 96, 96, 0, 1);
  conv(uf, nbr_full, P[31], nullptr, t2, N0V, 27, 96, 96);     // cat region is free now
  bn(t2, P[32], P[33], sf, yout, N0V, 96, 96, 0, 1);

  // ---- res4 (96 -> 96, identity skip) -> d_out y
  conv(yout, nbr_full, P[37], nullptr, tf, N0V, 27, 96, 96);
  bn(tf, P[38], P[39], nullptr, uf, N0V, 96, 96, 0, 1);
  conv(uf, nbr_full, P[40], nullptr, tf, N0V, 27, 96, 96);
  bn(tf, P[41], P[42], yout, yout, N0V, 96, 96, 0, 1);

  // ---- head: logits = y @ head_w + head_b
  conv(yout, nullptr, P[43], P[44], logits, N0V, 1, 96, 20);
}
